// RADModel_6253472383597
// MI455X (gfx1250) — matrix operand order repaired and checked by reading, not compiled, not run on hardware
//
#include <hip/hip_runtime.h>
#include <math.h>

#define T_SEQ 512
#define B_SZ  32
#define F_IN  16
#define HDIM  256
#define LATD  128
#define NMEM  65536
#define KSEL  16
#define NHEAD 8
#define HFD   128

typedef __attribute__((ext_vector_type(16))) _Float16 v16h;
typedef __attribute__((ext_vector_type(8)))  float    v8f;

__device__ __forceinline__ float gelu_f(float x) {
    return 0.5f * x * (1.f + erff(x * 0.70710678118654752f));
}

// ---------------------------------------------------------------------------
// f32 -> f16 weight conversion (lets recurrent scans load B fragments as
// single aligned 32-byte v16h vectors)
// ---------------------------------------------------------------------------
__global__ void f32_to_f16_kernel(const float* __restrict__ in,
                                  _Float16* __restrict__ out, int n) {
    int e = blockIdx.x * blockDim.x + threadIdx.x;
    if (e < n) out[e] = (_Float16)in[e];
}

// ---------------------------------------------------------------------------
// Small input projection: out(T,B,HD) = x(B,T,F) @ W(HD,F)^T + b   (K=16)
// ---------------------------------------------------------------------------
__global__ void proj16_kernel(const float* __restrict__ x,
                              const float* __restrict__ W,
                              const float* __restrict__ bias,
                              float* __restrict__ out) {
    long e = (long)blockIdx.x * blockDim.x + threadIdx.x;  // T*B*HD
    if (e >= (long)T_SEQ * B_SZ * HDIM) return;
    int o = (int)(e % HDIM);
    long tb = e / HDIM;
    int b = (int)(tb % B_SZ);
    int t = (int)(tb / B_SZ);
    const float* xp = x + ((long)b * T_SEQ + t) * F_IN;
    const float* wp = W + (long)o * F_IN;
    float s = bias[o];
#pragma unroll
    for (int f = 0; f < F_IN; ++f) s += xp[f] * wp[f];
    out[e] = s;
}

// ---------------------------------------------------------------------------
// Generic WMMA GEMM:  C(M,N) = act( A(M,K) * W(N,K)^T + bias ) [+ residual]
// One wave per 16x64 C strip (4 accumulators reusing one A fragment).
// grid = (N/64, M/16), block = 32 (wave32).  f16 operands, f32 accumulation.
// A and B fragments both use contiguous k order: k = 16*(lane/16) + s.
// ---------------------------------------------------------------------------
__global__ void wmma_gemm_nt(const float* __restrict__ A, int lda,
                             const float* __restrict__ W, int ldw,
                             const float* __restrict__ bias,
                             const float* __restrict__ residual, int ldr,
                             float* __restrict__ C, int ldc,
                             int K, int act) {
    const int ng = blockIdx.x, mt = blockIdx.y;
    const int lane = threadIdx.x & 31;
    const int rowA  = mt * 16 + (lane & 15);
    const int aKhi  = (lane & 16) ? 16 : 0;  // A: lanes 16-31 hold K+16 group
    const int bKhi  = (lane & 16) ? 16 : 0;  // B: lanes 16-31 hold K+16 group
    const int nLane = lane & 15;
    const int nBase = ng * 64;
    v8f acc[4] = {};
    const float* aB = A + (long)rowA * lda + aKhi;
    const float* wB[4];
#pragma unroll
    for (int t = 0; t < 4; ++t)
        wB[t] = W + (long)(nBase + t * 16 + nLane) * ldw + bKhi;
    for (int kb = 0; kb < K; kb += 32) {
        const float* ap = aB + kb;
        __builtin_prefetch((const void*)(wB[0] + kb + 128), 0, 3);  // global_prefetch_b8
        v16h a;
#pragma unroll
        for (int x2 = 0; x2 < 8; ++x2) a[x2] = (_Float16)ap[x2];
#pragma unroll
        for (int x2 = 0; x2 < 8; ++x2) a[8 + x2] = (_Float16)ap[8 + x2];
#pragma unroll
        for (int t = 0; t < 4; ++t) {
            const float* wp = wB[t] + kb;
            v16h b;
#pragma unroll
            for (int x2 = 0; x2 < 16; ++x2) b[x2] = (_Float16)wp[x2];
            acc[t] = __builtin_amdgcn_wmma_f32_16x16x32_f16(false, a, false, b,
                                                            (short)0, acc[t], false, false);
        }
    }
    const int rBase = mt * 16 + ((lane & 16) ? 8 : 0);
#pragma unroll
    for (int t = 0; t < 4; ++t) {
        const int col = nBase + t * 16 + nLane;
        const float bv = bias ? bias[col] : 0.f;
#pragma unroll
        for (int r = 0; r < 8; ++r) {
            const long row = rBase + r;
            float v = acc[t][r] + bv;
            if (act == 1) v = gelu_f(v);
            if (residual) v += residual[row * (long)ldr + col];
            C[row * (long)ldc + col] = v;
        }
    }
}

// ---------------------------------------------------------------------------
// Persistent GRU scan. One workgroup (8 waves) per direction; h lives in LDS.
// Per step:
//   1) async-stage this step's Gi row (32 x 3H f32) global->LDS
//      (GLOBAL_LOAD_ASYNC_TO_LDS_B128, tracked by ASYNCcnt) overlapped with
//   2) Gh = h @ Whh^T via WMMA (4 N-tiles per wave reusing one A fragment)
//   3) s_wait_asynccnt 0 + barrier, then gate math from LDS.
// Dynamic LDS: B*H (h) + B*3H (Gh) + B*3H (Gi stage) floats.
// ---------------------------------------------------------------------------
__global__ void gru_scan_kernel(const float* __restrict__ Gi0, long giDirStride,
                                const _Float16* __restrict__ W0, long wDirStride,
                                const float* __restrict__ bhh0, long bDirStride,
                                float* __restrict__ outY, int outW,
                                float* __restrict__ hT, int H) {
    extern __shared__ float smem[];
    float* h_lds  = smem;                        // B_SZ * H
    float* gh_lds = smem + B_SZ * H;             // B_SZ * 3H
    float* gi_lds = smem + B_SZ * 4 * H;         // B_SZ * 3H
    const unsigned giBaseOff = (unsigned)(B_SZ * 4 * H * sizeof(float));
    const int dir = blockIdx.x;
    const float*    Gi  = Gi0  + (long)dir * giDirStride;
    const _Float16* Whh = W0   + (long)dir * wDirStride;
    const float*    bhh = bhh0 + (long)dir * bDirStride;
    const int colOff = dir * H;
    const int tid  = threadIdx.x;
    const int lane = tid & 31;
    const int wave = tid >> 5;            // 0..7
    const int threeH  = 3 * H;
    const int kSteps  = H >> 5;           // 8 or 4
    const int nGroups = (threeH >> 4) >> 2;  // 12 (H=256) or 6 (H=128)
    const int totG    = 2 * nGroups;      // M=32 -> 2 m-tiles
    const int giVecs  = (B_SZ * threeH) >> 2;  // b128 transfers per step

    for (int e = tid; e < B_SZ * H; e += 256) h_lds[e] = 0.f;
    __syncthreads();

    for (int s = 0; s < T_SEQ; ++s) {
        const int tphys = dir ? (T_SEQ - 1 - s) : s;
        // ---- 1) async-stage Gi[tphys] into LDS (overlaps WMMA below) ----
        {
            const float* gsrc = Gi + (long)tphys * (B_SZ * threeH);
            for (int v = tid; v < giVecs; v += 256) {
                unsigned ldsOff = giBaseOff + ((unsigned)v << 4);
                const float* ga = gsrc + (v << 2);
                asm volatile("global_load_async_to_lds_b128 %0, %1, off"
                             :: "v"(ldsOff), "v"(ga) : "memory");
            }
        }
        // ---- 2) Gh = h @ Whh^T (WMMA, 16x64 strip per wave) ----
        for (int g = wave; g < totG; g += 8) {
            const int mt  = g & 1;
            const int ngi = g >> 1;
            v8f acc[4] = {};
            const int rowA = mt * 16 + (lane & 15);
            const int aKhi = (lane & 16) ? 16 : 0;
            const int bKhi = (lane & 16) ? 16 : 0;
            const int nB   = ngi * 64 + (lane & 15);
            for (int kk = 0; kk < kSteps; ++kk) {
                const int kb = kk * 32;
                const float* ap = h_lds + rowA * H + kb + aKhi;
                v16h a;
#pragma unroll
                for (int x2 = 0; x2 < 8; ++x2) a[x2] = (_Float16)ap[x2];
#pragma unroll
                for (int x2 = 0; x2 < 8; ++x2) a[8 + x2] = (_Float16)ap[8 + x2];
#pragma unroll
                for (int t = 0; t < 4; ++t) {
                    v16h b = *(const v16h*)(Whh + (long)(nB + t * 16) * H + kb + bKhi);
                    acc[t] = __builtin_amdgcn_wmma_f32_16x16x32_f16(false, a, false, b,
                                                                    (short)0, acc[t], false, false);
                }
            }
            const int rBase = mt * 16 + ((lane & 16) ? 8 : 0);
#pragma unroll
            for (int t = 0; t < 4; ++t) {
                const int col = ngi * 64 + t * 16 + (lane & 15);
#pragma unroll
                for (int r = 0; r < 8; ++r)
                    gh_lds[(rBase + r) * threeH + col] = acc[t][r];
            }
        }
        // ---- 3) wait async Gi stage, sync, gate math ----
        asm volatile("s_wait_asynccnt 0" ::: "memory");
        __syncthreads();
        for (int e = tid; e < B_SZ * H; e += 256) {
            const int b = e / H, j = e - b * H;
            const float gir = gi_lds[b * threeH + j];
            const float giz = gi_lds[b * threeH + H + j];
            const float gin = gi_lds[b * threeH + 2 * H + j];
            const float ghr = gh_lds[b * threeH + j]         + bhh[j];
            const float ghz = gh_lds[b * threeH + H + j]     + bhh[H + j];
            const float ghn = gh_lds[b * threeH + 2 * H + j] + bhh[2 * H + j];
            const float rg = 1.f / (1.f + __expf(-(gir + ghr)));
            const float zg = 1.f / (1.f + __expf(-(giz + ghz)));
            const float ng = tanhf(gin + rg * ghn);
            const float hnew = (1.f - zg) * ng + zg * h_lds[e];
            h_lds[e] = hnew;
            if (outY) outY[((long)tphys * B_SZ + b) * outW + colOff + j] = hnew;
        }
        __syncthreads();
    }
    if (hT && dir == 0) {
        for (int e = tid; e < B_SZ * H; e += 256) hT[e] = h_lds[e];
    }
}

// ---------------------------------------------------------------------------
// LayerNorm (+ optional exact GELU, + optional residual add). 1 block / row.
// ---------------------------------------------------------------------------
__global__ void ln_kernel(const float* __restrict__ in,
                          const float* __restrict__ g, const float* __restrict__ b,
                          const float* __restrict__ res,
                          float* __restrict__ out, int width, int doGelu) {
    const int row = blockIdx.x, tid = threadIdx.x;
    __shared__ float red[256];
    const float* p = in + (long)row * width;
    float s = 0.f;
    for (int j = tid; j < width; j += 256) s += p[j];
    red[tid] = s; __syncthreads();
    for (int st = 128; st > 0; st >>= 1) { if (tid < st) red[tid] += red[tid + st]; __syncthreads(); }
    const float mean = red[0] / width;
    __syncthreads();
    float v = 0.f;
    for (int j = tid; j < width; j += 256) { float d = p[j] - mean; v += d * d; }
    red[tid] = v; __syncthreads();
    for (int st = 128; st > 0; st >>= 1) { if (tid < st) red[tid] += red[tid + st]; __syncthreads(); }
    const float inv = rsqrtf(red[0] / width + 1e-5f);
    __syncthreads();
    for (int j = tid; j < width; j += 256) {
        float val = (p[j] - mean) * inv * g[j] + b[j];
        if (doGelu) val = gelu_f(val);
        if (res) val += res[(long)row * width + j];
        out[(long)row * width + j] = val;
    }
}

// L2-normalize rows in place (width=128), max(norm,1e-12)
__global__ void l2norm_kernel(float* __restrict__ v, int width) {
    const int row = blockIdx.x, tid = threadIdx.x;  // 128 threads
    __shared__ float red[128];
    float* p = v + (long)row * width;
    float x = p[tid];
    red[tid] = x * x; __syncthreads();
    for (int st = 64; st > 0; st >>= 1) { if (tid < st) red[tid] += red[tid + st]; __syncthreads(); }
    const float nrm = fmaxf(sqrtf(red[0]), 1e-12f);
    p[tid] = x / nrm;
}

// (T,B,HD) -> (B,T,HD)
__global__ void tb_transpose_kernel(const float* __restrict__ in,
                                    float* __restrict__ out) {
    long e = (long)blockIdx.x * blockDim.x + threadIdx.x;
    if (e >= (long)T_SEQ * B_SZ * HDIM) return;
    int c = (int)(e % HDIM);
    long tb = e / HDIM;
    int b = (int)(tb % B_SZ);
    int t = (int)(tb / B_SZ);
    out[((long)b * T_SEQ + t) * HDIM + c] = in[e];
}

// ---------------------------------------------------------------------------
// Masked top-k (k=16) with SOH relaxation. One block (256 thr) per batch row.
// ---------------------------------------------------------------------------
__global__ void topk_kernel(const float* __restrict__ sim,     // (B, NMEM)
                            const float* __restrict__ soh,     // (B)
                            const float* __restrict__ memsoh,  // (NMEM)
                            int* __restrict__ idxOut) {        // (B, 16)
    const int b = blockIdx.x, tid = threadIdx.x;
    __shared__ int   scnt[256];
    __shared__ float cval[256 * KSEL];
    __shared__ int   cidx[256 * KSEL];
    __shared__ float rval[256];
    __shared__ int   rpos[256];
    const float sc = soh[b];
    int cnt = 0;
    for (int i = tid; i < NMEM; i += 256)
        if (fabsf(sc - memsoh[i]) <= 0.05f) cnt++;
    scnt[tid] = cnt; __syncthreads();
    for (int st = 128; st > 0; st >>= 1) { if (tid < st) scnt[tid] += scnt[tid + st]; __syncthreads(); }
    const float tol = (scnt[0] < KSEL) ? 0.10f : 0.05f;
    __syncthreads();
    float lv[KSEL]; int li[KSEL];
#pragma unroll
    for (int r = 0; r < KSEL; ++r) { lv[r] = -INFINITY; li[r] = 0; }
    const float* srow = sim + (long)b * NMEM;
    for (int i = tid; i < NMEM; i += 256) {
        float sv = (fabsf(sc - memsoh[i]) <= tol) ? srow[i] : -INFINITY;
        if (sv > lv[KSEL - 1]) {
            int p2 = KSEL - 1;
            while (p2 > 0 && lv[p2 - 1] < sv) { lv[p2] = lv[p2 - 1]; li[p2] = li[p2 - 1]; --p2; }
            lv[p2] = sv; li[p2] = i;
        }
    }
#pragma unroll
    for (int r = 0; r < KSEL; ++r) { cval[tid * KSEL + r] = lv[r]; cidx[tid * KSEL + r] = li[r]; }
    __syncthreads();
    for (int sel = 0; sel < KSEL; ++sel) {
        float best = -INFINITY; int bp = -1;
#pragma unroll
        for (int r = 0; r < KSEL; ++r) {
            float vv = cval[tid * KSEL + r];
            if (vv > best) { best = vv; bp = tid * KSEL + r; }
        }
        rval[tid] = best; rpos[tid] = bp; __syncthreads();
        for (int st = 128; st > 0; st >>= 1) {
            if (tid < st && rval[tid + st] > rval[tid]) { rval[tid] = rval[tid + st]; rpos[tid] = rpos[tid + st]; }
            __syncthreads();
        }
        if (tid == 0) {
            int w = rpos[0];
            idxOut[b * KSEL + sel] = (w >= 0) ? cidx[w] : 0;
            if (w >= 0) cval[w] = -INFINITY;
        }
        __syncthreads();
    }
}

// retrieved(B*K, LAT) = mem_keys[idx]
__global__ void gather_kernel(const float* __restrict__ mem,
                              const int* __restrict__ idx,
                              float* __restrict__ out) {
    long e = (long)blockIdx.x * blockDim.x + threadIdx.x;
    if (e >= (long)B_SZ * KSEL * LATD) return;
    int d = (int)(e % LATD);
    int r = (int)(e / LATD);
    out[e] = mem[(long)idx[r] * LATD + d];
}

// ---------------------------------------------------------------------------
// Cross-attention over k=16 slots. One block per (b, head); K/V head tile in LDS.
// ---------------------------------------------------------------------------
__global__ void attn_kernel(const float* __restrict__ Q,    // (B*T, HD)
                            const float* __restrict__ KV,   // (B*K, 2*HD): [k | v]
                            float* __restrict__ O) {        // (B*T, HD)
    const int bh = blockIdx.x;
    const int b = bh / NHEAD, h = bh % NHEAD;
    const int tid = threadIdx.x;
    __shared__ float kv_s[KSEL * 64];  // [j][0..31]=k, [j][32..63]=v
    for (int e = tid; e < KSEL * 64; e += 256) {
        int j = e >> 6, d = e & 63;
        int col = (d < 32) ? (h * 32 + d) : (HDIM + h * 32 + (d - 32));
        kv_s[e] = KV[((long)b * KSEL + j) * (2 * HDIM) + col];
    }
    __syncthreads();
    const float scale = 0.17677669529663687f;  // 1/sqrt(32)
    for (int t = tid; t < T_SEQ; t += 256) {
        const float* qp = Q + ((long)b * T_SEQ + t) * HDIM + h * 32;
        float q[32];
#pragma unroll
        for (int d = 0; d < 32; ++d) q[d] = qp[d];
        float sv[KSEL]; float mx = -INFINITY;
#pragma unroll
        for (int j = 0; j < KSEL; ++j) {
            float acc = 0.f;
#pragma unroll
            for (int d = 0; d < 32; ++d) acc += q[d] * kv_s[j * 64 + d];
            sv[j] = acc * scale;
            mx = fmaxf(mx, sv[j]);
        }
        float den = 0.f;
#pragma unroll
        for (int j = 0; j < KSEL; ++j) { sv[j] = __expf(sv[j] - mx); den += sv[j]; }
        const float inv = 1.f / den;
        float o[32];
#pragma unroll
        for (int d = 0; d < 32; ++d) o[d] = 0.f;
#pragma unroll
        for (int j = 0; j < KSEL; ++j) {
            float pj = sv[j] * inv;
#pragma unroll
            for (int d = 0; d < 32; ++d) o[d] += pj * kv_s[j * 64 + 32 + d];
        }
        float* op = O + ((long)b * T_SEQ + t) * HDIM + h * 32;
#pragma unroll
        for (int d = 0; d < 32; ++d) op[d] = o[d];
    }
}

// ---------------------------------------------------------------------------
extern "C" void kernel_launch(void* const* d_in, const int* in_sizes, int n_in,
                              void* d_out, int out_size, void* d_ws, size_t ws_size,
                              hipStream_t stream) {
    (void)in_sizes; (void)n_in; (void)out_size; (void)ws_size;
    const float* x         = (const float*)d_in[0];
    const float* soh       = (const float*)d_in[1];
    const float* mem_keys  = (const float*)d_in[2];
    const float* mem_soh   = (const float*)d_in[3];
    const float* enc_in_w  = (const float*)d_in[4];
    const float* enc_in_b  = (const float*)d_in[5];
    const float* egru_w_ih = (const float*)d_in[6];
    const float* egru_w_hh = (const float*)d_in[7];
    const float* egru_b_ih = (const float*)d_in[8];
    const float* egru_b_hh = (const float*)d_in[9];
    const float* lp1_w = (const float*)d_in[10];
    const float* lp1_b = (const float*)d_in[11];
    const float* ln1_g = (const float*)d_in[12];
    const float* ln1_b = (const float*)d_in[13];
    const float* lp2_w = (const float*)d_in[14];
    const float* lp2_b = (const float*)d_in[15];
    const float* fp_w  = (const float*)d_in[16];
    const float* fp_b  = (const float*)d_in[17];
    const float* fgru_w_ih = (const float*)d_in[18];
    const float* fgru_w_hh = (const float*)d_in[19];
    const float* fgru_b_ih = (const float*)d_in[20];
    const float* fgru_b_hh = (const float*)d_in[21];
    const float* rp_w = (const float*)d_in[22];
    const float* rp_b = (const float*)d_in[23];
    const float* attn_in_w  = (const float*)d_in[24];
    const float* attn_in_b  = (const float*)d_in[25];
    const float* attn_out_w = (const float*)d_in[26];
    const float* attn_out_b = (const float*)d_in[27];
    const float* ffn1_w = (const float*)d_in[28];
    const float* ffn1_b = (const float*)d_in[29];
    const float* ffn2_w = (const float*)d_in[30];
    const float* ffn2_b = (const float*)d_in[31];
    const float* ffn_ln_g = (const float*)d_in[32];
    const float* ffn_ln_b = (const float*)d_in[33];

    float* ws   = (float*)d_ws;
    float* BUF0 = ws;                    // (T*B,256): seq_e / inp_f / Q
    float* BUF1 = ws + 4194304;          // ys_e / feat-l1-out / O
    float* BUF2 = ws + 8388608;          // feat-l0-out / fused
    float* BUF3 = ws + 12582912;         // feats (B,T,HD) / ff2
    float* BIG  = ws + 16777216;         // Gi (16384x768) / sim / FFN-hidden
    float* HT   = ws + 29360128;         // (32,256)
    float* A1   = HT + 8192;
    float* A2   = A1 + 8192;
    float* LAT0 = A2 + 8192;             // (32,128)
    float* RETR = LAT0 + 4096;           // (512,128)
    float* RPB  = RETR + 65536;          // (512,256)
    float* KVB  = RPB + 131072;          // (512,512)
    int*   IDX  = (int*)(KVB + 262144);  // (32,16)
    _Float16* WH16E = (_Float16*)(ws + 29848064);      // egru w_hh f16 (2x768x256)
    _Float16* WH16F = WH16E + 2 * 768 * 256;           // fgru w_hh f16 (4x384x128)

    const dim3 blk32(32), blk128(128), blk256(256);
    const size_t smemE = (size_t)(B_SZ * 7 * HDIM) * sizeof(float);  // h + Gh + Gi stage
    const size_t smemF = (size_t)(B_SZ * 7 * HFD)  * sizeof(float);

    // 0) convert recurrent weights to f16 once
    f32_to_f16_kernel<<<dim3((2 * 768 * 256 + 255) / 256), blk256, 0, stream>>>(egru_w_hh, WH16E, 2 * 768 * 256);
    f32_to_f16_kernel<<<dim3((4 * 384 * 128 + 255) / 256), blk256, 0, stream>>>(fgru_w_hh, WH16F, 4 * 384 * 128);

    // ---- Encoder ----
    proj16_kernel<<<dim3(16384), blk256, 0, stream>>>(x, enc_in_w, enc_in_b, BUF0);
    wmma_gemm_nt<<<dim3(12, 1024), blk32, 0, stream>>>(BUF0, HDIM, egru_w_ih, HDIM, egru_b_ih, nullptr, 0, BIG, 768, HDIM, 0);
    gru_scan_kernel<<<dim3(1), blk256, smemE, stream>>>(BIG, 0, WH16E, 0, egru_b_hh, 0, BUF1, HDIM, nullptr, HDIM);
    wmma_gemm_nt<<<dim3(12, 1024), blk32, 0, stream>>>(BUF1, HDIM, egru_w_ih + 768 * 256, HDIM, egru_b_ih + 768, nullptr, 0, BIG, 768, HDIM, 0);
    gru_scan_kernel<<<dim3(1), blk256, smemE, stream>>>(BIG, 0, WH16E + 768 * 256, 0, egru_b_hh + 768, 0, nullptr, HDIM, HT, HDIM);
    wmma_gemm_nt<<<dim3(4, 2), blk32, 0, stream>>>(HT, HDIM, lp1_w, HDIM, lp1_b, nullptr, 0, A1, HDIM, HDIM, 0);
    ln_kernel<<<dim3(32), blk256, 0, stream>>>(A1, ln1_g, ln1_b, nullptr, A2, HDIM, 1);
    wmma_gemm_nt<<<dim3(2, 2), blk32, 0, stream>>>(A2, HDIM, lp2_w, HDIM, lp2_b, nullptr, 0, LAT0, LATD, HDIM, 0);
    l2norm_kernel<<<dim3(32), blk128, 0, stream>>>(LAT0, LATD);

    // ---- Feature path: 2-layer bidirectional GRU ----
    proj16_kernel<<<dim3(16384), blk256, 0, stream>>>(x, fp_w, fp_b, BUF0);
    wmma_gemm_nt<<<dim3(6, 1024), blk32, 0, stream>>>(BUF0, HDIM, fgru_w_ih, HDIM, fgru_b_ih, nullptr, 0, BIG, 384, HDIM, 0);
    wmma_gemm_nt<<<dim3(6, 1024), blk32, 0, stream>>>(BUF0, HDIM, fgru_w_ih + 384 * 256, HDIM, fgru_b_ih + 384, nullptr, 0, BIG + 6291456, 384, HDIM, 0);
    gru_scan_kernel<<<dim3(2), blk256, smemF, stream>>>(BIG, 6291456, WH16F, 384 * 128, fgru_b_hh, 384, BUF2, HDIM, nullptr, HFD);
    wmma_gemm_nt<<<dim3(6, 1024), blk32, 0, stream>>>(BUF2, HDIM, fgru_w_ih + 2 * 384 * 256, HDIM, fgru_b_ih + 2 * 384, nullptr, 0, BIG, 384, HDIM, 0);
    wmma_gemm_nt<<<dim3(6, 1024), blk32, 0, stream>>>(BUF2, HDIM, fgru_w_ih + 3 * 384 * 256, HDIM, fgru_b_ih + 3 * 384, nullptr, 0, BIG + 6291456, 384, HDIM, 0);
    gru_scan_kernel<<<dim3(2), blk256, smemF, stream>>>(BIG, 6291456, WH16F + 2 * 384 * 128, 384 * 128, fgru_b_hh + 2 * 384, 384, BUF1, HDIM, nullptr, HFD);
    tb_transpose_kernel<<<dim3(16384), blk256, 0, stream>>>(BUF1, BUF3);

    // ---- Memory retrieval ----
    wmma_gemm_nt<<<dim3(1024, 2), blk32, 0, stream>>>(LAT0, LATD, mem_keys, LATD, nullptr, nullptr, 0, BIG, NMEM, LATD, 0);
    topk_kernel<<<dim3(32), blk256, 0, stream>>>(BIG, soh, mem_soh, IDX);
    gather_kernel<<<dim3(256), blk256, 0, stream>>>(mem_keys, IDX, RETR);
    wmma_gemm_nt<<<dim3(4, 32), blk32, 0, stream>>>(RETR, LATD, rp_w, LATD, rp_b, nullptr, 0, RPB, HDIM, LATD, 0);

    // ---- Attention fusion ----
    wmma_gemm_nt<<<dim3(4, 1024), blk32, 0, stream>>>(BUF3, HDIM, attn_in_w, HDIM, attn_in_b, nullptr, 0, BUF0, HDIM, HDIM, 0);
    wmma_gemm_nt<<<dim3(8, 32), blk32, 0, stream>>>(RPB, HDIM, attn_in_w + 256 * 256, HDIM, attn_in_b + 256, nullptr, 0, KVB, 512, HDIM, 0);
    attn_kernel<<<dim3(256), blk256, 0, stream>>>(BUF0, KVB, BUF1);
    wmma_gemm_nt<<<dim3(4, 1024), blk32, 0, stream>>>(BUF1, HDIM, attn_out_w, HDIM, attn_out_b, BUF3, HDIM, BUF2, HDIM, HDIM, 0);

    // ---- FFN + final LN + residual ----
    wmma_gemm_nt<<<dim3(8, 1024), blk32, 0, stream>>>(BUF2, HDIM, ffn1_w, HDIM, ffn1_b, nullptr, 0, BIG, 512, HDIM, 1);
    wmma_gemm_nt<<<dim3(4, 1024), blk32, 0, stream>>>(BIG, 512, ffn2_w, 512, ffn2_b, nullptr, 0, BUF3, HDIM, 512, 0);
    ln_kernel<<<dim3(16384), blk256, 0, stream>>>(BUF3, ffn_ln_g, ffn_ln_b, BUF2, (float*)d_out, HDIM, 0);
}
